// MolecularGraphEncoder_14070312862352
// MI455X (gfx1250) — compile-verified, matrix operand order checked by reading, not run on hardware
//
#include <hip/hip_runtime.h>

// Problem dims (fixed by the reference)
#define N_NODES 262144
#define N_EDGES 1048576
#define N_GRAPH 8192
#define ND 32
#define ED 10
#define H 96
#define OUT_D 256
#define NLAYER 3
#define EPS_C 1e-5f

typedef __attribute__((ext_vector_type(16))) __bf16 v16bf;
typedef __attribute__((ext_vector_type(8)))  float  v8f;

// ---- bf16 helpers (RNE float->bf16, stored as ushort to avoid __bf16 arith) ----
__device__ __forceinline__ unsigned short f2bf(float f) {
  unsigned int u = __float_as_uint(f);
  u += 0x7FFFu + ((u >> 16) & 1u);
  return (unsigned short)(u >> 16);
}
__device__ __forceinline__ unsigned long long pack4bf(float a, float b, float c, float d) {
  return (unsigned long long)f2bf(a)
       | ((unsigned long long)f2bf(b) << 16)
       | ((unsigned long long)f2bf(c) << 32)
       | ((unsigned long long)f2bf(d) << 48);
}

// A operand: 16xK bf16 tile in LDS, row-major with stride ld (elements).
// ISA layout (16-bit A 16x32): lane<16 holds row=lane, K in {kb0+0..7, kb0+16..23};
// lane>=16 holds same row, K offset +8. Both halves are 16B-contiguous -> 2x ds_load_b128.
__device__ __forceinline__ v16bf load_a_frag(const unsigned short* A, int ld, int kb0, int lane) {
  int row  = lane & 15;
  int koff = kb0 + ((lane >> 4) << 3);
  v16bf r;
  __builtin_memcpy(&r, A + row * ld + koff, 16);
  __builtin_memcpy(((char*)&r) + 16, A + row * ld + koff + 16, 16);
  return r;
}

// Stage KxNcols f32 weights from global into LDS in *fragment order*:
// Bsw[chunk][nt][lane][16], so a lane's whole B operand is 32 contiguous bytes.
__device__ __forceinline__ void stage_b_swizzled(const float* __restrict__ W, int ldb,
                                                 int nchunk, int nnt,
                                                 unsigned short* Bsw, int tid, int nthreads) {
  const int total = nchunk * nnt * 32 * 16;
  for (int idx = tid; idx < total; idx += nthreads) {
    int j    = idx & 15;
    int lane = (idx >> 4) & 31;
    int nt   = (idx >> 9) % nnt;
    int ch   = idx / (nnt * 512);
    int k    = ch * 32 + ((lane >> 4) << 3) + ((j >> 3) << 4) + (j & 7);
    int col  = nt * 16 + (lane & 15);
    Bsw[idx] = f2bf(W[k * ldb + col]);
  }
}
__device__ __forceinline__ v16bf load_b_frag_sw(const unsigned short* Bsw, int nnt,
                                                int ch, int nt, int lane) {
  const unsigned short* p = Bsw + (((ch * nnt + nt) * 32 + lane) << 4);
  v16bf r; __builtin_memcpy(&r, p, 32);   // 2x ds_load_b128
  return r;
}
// B operand straight from global f32 weights (head GEMM2 only; L2-resident)
__device__ __forceinline__ v16bf load_b_frag_g(const float* B, int ldb, int kb0, int nb, int lane) {
  int col  = nb + (lane & 15);
  int koff = kb0 + ((lane >> 4) << 3);
  unsigned short tmp[16];
#pragma unroll
  for (int j = 0; j < 8; ++j) {
    tmp[j]     = f2bf(B[(koff + j) * ldb + col]);
    tmp[8 + j] = f2bf(B[(koff + 16 + j) * ldb + col]);
  }
  v16bf r; __builtin_memcpy(&r, tmp, 32); return r;
}
__device__ __forceinline__ v8f wmma_bf16(v16bf a, v16bf b, v8f c) {
  // (neg_a, A, neg_b, B, c_mod, C, reuse_a, reuse_b)
  return __builtin_amdgcn_wmma_f32_16x16x32_bf16(false, a, false, b, (short)0, c, false, false);
}

// =================== small utility kernels ===================
__global__ void k_count(const int* __restrict__ batch, float* __restrict__ cnt) {
  int n = blockIdx.x * blockDim.x + threadIdx.x;
  if (n < N_NODES) atomicAdd(&cnt[batch[n]], 1.0f);
}
__global__ void k_inv(const float* __restrict__ cnt, float* __restrict__ inv) {
  int g = blockIdx.x * blockDim.x + threadIdx.x;
  if (g < N_GRAPH) inv[g] = 1.0f / fmaxf(cnt[g], 1.0f);
}

// =================== node embedding: h = relu(x @ node_W + b), WMMA K=32 ===================
#define EMB_WAVES 8
__global__ void __launch_bounds__(256) k_embed(const float* __restrict__ x,
                                               const float* __restrict__ W,
                                               const float* __restrict__ b,
                                               float* __restrict__ hout, int ntiles) {
  __shared__ __align__(16) unsigned short Ws[ND * H];              // 1 chunk x 6 nt, swizzled
  __shared__ __align__(16) unsigned short Ain[EMB_WAVES][16 * ND];
  const int tid = threadIdx.x, lane = tid & 31, w = tid >> 5;
  stage_b_swizzled(W, H, 1, 6, Ws, tid, blockDim.x);
  __syncthreads();
  const int nw = gridDim.x * EMB_WAVES;
  for (int tile = blockIdx.x * EMB_WAVES + w; tile < ntiles; tile += nw) {
    const int n0 = tile * 16;
    for (int i = lane * 4; i < 16 * ND; i += 128) {
      const float4 v = *(const float4*)(x + n0 * ND + i);
      *(unsigned long long*)(&Ain[w][i]) = pack4bf(v.x, v.y, v.z, v.w);
    }
    __syncthreads();
    const int crow = (lane >> 4) << 3, col = lane & 15;
#pragma unroll
    for (int nt = 0; nt < 6; ++nt) {
      v8f acc;
      const float bias = b[nt * 16 + col];
#pragma unroll
      for (int r = 0; r < 8; ++r) acc[r] = bias;
      v16bf a  = load_a_frag(&Ain[w][0], ND, 0, lane);
      v16bf bb = load_b_frag_sw(Ws, 6, 0, nt, lane);
      acc = wmma_bf16(a, bb, acc);
#pragma unroll
      for (int r = 0; r < 8; ++r)
        hout[(n0 + r + crow) * H + nt * 16 + col] = fmaxf(acc[r], 0.0f);
    }
    __syncthreads();
  }
}

// ===== edge pass (4 channels/thread): agg[dst] += relu(h[src] + relu(edge_attr@edge_W + b)) =====
__global__ void __launch_bounds__(256) k_edge(const float* __restrict__ h,
                                              const float* __restrict__ eattr,
                                              const int* __restrict__ src,
                                              const int* __restrict__ dst,
                                              const float* __restrict__ eW,
                                              const float* __restrict__ eb,
                                              float* __restrict__ agg) {
  int idx = blockIdx.x * blockDim.x + threadIdx.x;        // E * H/4 threads
  if (idx >= N_EDGES * (H / 4)) return;
  int e  = idx / (H / 4);
  int c0 = (idx - e * (H / 4)) * 4;
  float ea[ED];
#pragma unroll
  for (int k = 0; k < ED; ++k) ea[k] = eattr[e * ED + k];
  float s[4];
#pragma unroll
  for (int q = 0; q < 4; ++q) s[q] = eb[c0 + q];
#pragma unroll
  for (int k = 0; k < ED; ++k) {
    const float4 wv = *(const float4*)(eW + k * H + c0);
    s[0] += ea[k] * wv.x; s[1] += ea[k] * wv.y; s[2] += ea[k] * wv.z; s[3] += ea[k] * wv.w;
  }
  const int sb = src[e] * H + c0, db = dst[e] * H + c0;
  const float4 hv = *(const float4*)(h + sb);
  float m0 = fmaxf(hv.x + fmaxf(s[0], 0.0f), 0.0f);
  float m1 = fmaxf(hv.y + fmaxf(s[1], 0.0f), 0.0f);
  float m2 = fmaxf(hv.z + fmaxf(s[2], 0.0f), 0.0f);
  float m3 = fmaxf(hv.w + fmaxf(s[3], 0.0f), 0.0f);
  atomicAdd(&agg[db + 0], m0);
  atomicAdd(&agg[db + 1], m1);
  atomicAdd(&agg[db + 2], m2);
  atomicAdd(&agg[db + 3], m3);
}

// ===== GINE MLP: t = relu(((1+eps)h + agg) @ W1 + b1) @ W2 + b2 ; fused double WMMA GEMM =====
#define MLP_WAVES 4
__global__ void __launch_bounds__(128) k_mlp(const float* __restrict__ h,
                                             const float* __restrict__ agg,
                                             const float* __restrict__ W1, const float* __restrict__ b1,
                                             const float* __restrict__ W2, const float* __restrict__ b2,
                                             const float* __restrict__ eps, int layer,
                                             float* __restrict__ t, int ntiles) {
  __shared__ __align__(16) unsigned short W1s[H * H];   // 3 chunks x 6 nt, fragment-swizzled
  __shared__ __align__(16) unsigned short W2s[H * H];
  __shared__ __align__(16) unsigned short Ain[MLP_WAVES][16 * H];
  __shared__ __align__(16) unsigned short Amid[MLP_WAVES][16 * H];
  const int tid = threadIdx.x, lane = tid & 31, w = tid >> 5;
  stage_b_swizzled(W1, H, 3, 6, W1s, tid, blockDim.x);
  stage_b_swizzled(W2, H, 3, 6, W2s, tid, blockDim.x);
  const float ev = 1.0f + eps[layer];
  __syncthreads();
  const int nw = gridDim.x * MLP_WAVES;  // grid chosen so ntiles % nw == 0 (uniform syncs)
  for (int tile = blockIdx.x * MLP_WAVES + w; tile < ntiles; tile += nw) {
    const int n0 = tile * 16;
    for (int i = lane * 4; i < 16 * H; i += 128) {
      const float4 hv = *(const float4*)(h + n0 * H + i);
      const float4 av = *(const float4*)(agg + n0 * H + i);
      *(unsigned long long*)(&Ain[w][i]) =
          pack4bf(ev * hv.x + av.x, ev * hv.y + av.y, ev * hv.z + av.z, ev * hv.w + av.w);
    }
    { int tn = tile + nw;
      if (tn < ntiles) __builtin_prefetch(&h[tn * 16 * H + lane * 48], 0, 1); }
    __syncthreads();
    const int crow = (lane >> 4) << 3, col = lane & 15;
#pragma unroll
    for (int nt = 0; nt < 6; ++nt) {
      v8f acc; const float bias = b1[nt * 16 + col];
#pragma unroll
      for (int r = 0; r < 8; ++r) acc[r] = bias;
#pragma unroll
      for (int kc = 0; kc < 3; ++kc) {
        v16bf a = load_a_frag(&Ain[w][0], H, kc * 32, lane);
        v16bf b = load_b_frag_sw(W1s, 6, kc, nt, lane);
        acc = wmma_bf16(a, b, acc);
      }
#pragma unroll
      for (int r = 0; r < 8; ++r)
        Amid[w][(r + crow) * H + nt * 16 + col] = f2bf(fmaxf(acc[r], 0.0f));
    }
    __syncthreads();
#pragma unroll
    for (int nt = 0; nt < 6; ++nt) {
      v8f acc; const float bias = b2[nt * 16 + col];
#pragma unroll
      for (int r = 0; r < 8; ++r) acc[r] = bias;
#pragma unroll
      for (int kc = 0; kc < 3; ++kc) {
        v16bf a = load_a_frag(&Amid[w][0], H, kc * 32, lane);
        v16bf b = load_b_frag_sw(W2s, 6, kc, nt, lane);
        acc = wmma_bf16(a, b, acc);
      }
#pragma unroll
      for (int r = 0; r < 8; ++r)
        t[(n0 + r + crow) * H + nt * 16 + col] = acc[r];
    }
    __syncthreads();
  }
}

// =================== GraphNorm kernels (4 channels/thread) ===================
__global__ void k_gsum(const float* __restrict__ t, const int* __restrict__ batch,
                       float* __restrict__ mean) {
  int idx = blockIdx.x * blockDim.x + threadIdx.x;      // N * H/4
  if (idx >= N_NODES * (H / 4)) return;
  int n = idx / (H / 4), c0 = (idx - n * (H / 4)) * 4;
  const float4 v = *(const float4*)(t + n * H + c0);
  float* m = mean + batch[n] * H + c0;
  atomicAdd(m + 0, v.x); atomicAdd(m + 1, v.y); atomicAdd(m + 2, v.z); atomicAdd(m + 3, v.w);
}
__global__ void k_gscale(float* __restrict__ mean, const float* __restrict__ inv) {
  int idx = blockIdx.x * blockDim.x + threadIdx.x;
  if (idx < N_GRAPH * H) mean[idx] *= inv[idx / H];
}
__global__ void k_gvar(const float* __restrict__ t, const int* __restrict__ batch,
                       const float* __restrict__ mean, const float* __restrict__ ga,
                       float* __restrict__ var) {
  int idx = blockIdx.x * blockDim.x + threadIdx.x;
  if (idx >= N_NODES * (H / 4)) return;
  int n = idx / (H / 4), c0 = (idx - n * (H / 4)) * 4, g = batch[n];
  const float4 v = *(const float4*)(t + n * H + c0);
  const float4 mv = *(const float4*)(mean + g * H + c0);
  const float4 av = *(const float4*)(ga + c0);
  float c0v = v.x - av.x * mv.x, c1v = v.y - av.y * mv.y;
  float c2v = v.z - av.z * mv.z, c3v = v.w - av.w * mv.w;
  float* vp = var + g * H + c0;
  atomicAdd(vp + 0, c0v * c0v); atomicAdd(vp + 1, c1v * c1v);
  atomicAdd(vp + 2, c2v * c2v); atomicAdd(vp + 3, c3v * c3v);
}
__global__ void k_gapply(const float* __restrict__ t, const int* __restrict__ batch,
                         const float* __restrict__ mean, const float* __restrict__ var,
                         const float* __restrict__ inv, const float* __restrict__ gw,
                         const float* __restrict__ gb, const float* __restrict__ ga,
                         float* __restrict__ h) {
  int idx = blockIdx.x * blockDim.x + threadIdx.x;
  if (idx >= N_NODES * (H / 4)) return;
  int n = idx / (H / 4), c0 = (idx - n * (H / 4)) * 4, g = batch[n];
  const float iv = inv[g];
  const float4 v  = *(const float4*)(t + n * H + c0);
  const float4 mv = *(const float4*)(mean + g * H + c0);
  const float4 vv = *(const float4*)(var + g * H + c0);
  const float4 av = *(const float4*)(ga + c0);
  const float4 wv = *(const float4*)(gw + c0);
  const float4 bv = *(const float4*)(gb + c0);
  float4 ho = *(float4*)(h + n * H + c0);
  ho.x += fmaxf(wv.x * (v.x - av.x * mv.x) * rsqrtf(vv.x * iv + EPS_C) + bv.x, 0.0f);
  ho.y += fmaxf(wv.y * (v.y - av.y * mv.y) * rsqrtf(vv.y * iv + EPS_C) + bv.y, 0.0f);
  ho.z += fmaxf(wv.z * (v.z - av.z * mv.z) * rsqrtf(vv.z * iv + EPS_C) + bv.z, 0.0f);
  ho.w += fmaxf(wv.w * (v.w - av.w * mv.w) * rsqrtf(vv.w * iv + EPS_C) + bv.w, 0.0f);
  *(float4*)(h + n * H + c0) = ho;   // relu + residual (h stays >= 0)
}

// ======== pooling: per-graph mean (atomicAdd) and max (int-bit atomicMax; h >= 0) ========
__global__ void k_pool(const float* __restrict__ h, const int* __restrict__ batch,
                       float* __restrict__ gmean, unsigned int* __restrict__ gmaxb) {
  int idx = blockIdx.x * blockDim.x + threadIdx.x;
  if (idx >= N_NODES * (H / 4)) return;
  int n = idx / (H / 4), c0 = (idx - n * (H / 4)) * 4, g = batch[n];
  const float4 v = *(const float4*)(h + n * H + c0);
  float* mp = gmean + g * H + c0;
  int*   xp = (int*)(gmaxb + g * H + c0);
  atomicAdd(mp + 0, v.x); atomicAdd(mp + 1, v.y); atomicAdd(mp + 2, v.z); atomicAdd(mp + 3, v.w);
  atomicMax(xp + 0, __float_as_int(v.x)); atomicMax(xp + 1, __float_as_int(v.y));
  atomicMax(xp + 2, __float_as_int(v.z)); atomicMax(xp + 3, __float_as_int(v.w));
}

// ======== rep = LayerNorm([mean || max]) per graph, 192 threads/graph ========
__global__ void __launch_bounds__(192) k_rep_ln(const float* __restrict__ gmean,
                                                const unsigned int* __restrict__ gmaxb,
                                                const float* __restrict__ inv,
                                                const float* __restrict__ lw,
                                                const float* __restrict__ lb,
                                                float* __restrict__ rep) {
  __shared__ float sv[2 * H];
  int g = blockIdx.x, c = threadIdx.x;
  float v = (c < H) ? gmean[g * H + c] * inv[g] : __uint_as_float(gmaxb[g * H + (c - H)]);
  sv[c] = v;
  __syncthreads();
  float s1 = 0.0f, s2 = 0.0f;
  for (int j = 0; j < 2 * H; ++j) { float x = sv[j]; s1 += x; s2 += x * x; }
  float mu  = s1 * (1.0f / (2 * H));
  float var = s2 * (1.0f / (2 * H)) - mu * mu;
  rep[g * 2 * H + c] = lw[c] * (v - mu) * rsqrtf(var + EPS_C) + lb[c];
}

// ======== head: out = relu(rep @ W1 + b1) @ W2 + b2 ; WMMA K=192 then K=96, OUT=256 ========
#define HEAD_WAVES 2
__global__ void __launch_bounds__(64) k_head(const float* __restrict__ rep,
                                             const float* __restrict__ W1, const float* __restrict__ b1,
                                             const float* __restrict__ W2, const float* __restrict__ b2,
                                             float* __restrict__ out) {
  __shared__ __align__(16) unsigned short W1s[2 * H * H];            // 6 chunks x 6 nt, swizzled
  __shared__ __align__(16) unsigned short Arep[HEAD_WAVES][16 * 2 * H];
  __shared__ __align__(16) unsigned short Amid[HEAD_WAVES][16 * H];
  const int tid = threadIdx.x, lane = tid & 31, w = tid >> 5;
  stage_b_swizzled(W1, H, 6, 6, W1s, tid, blockDim.x);
  __syncthreads();
  const int tile = blockIdx.x * HEAD_WAVES + w;          // G/16 tiles, 1 per wave
  const int g0 = tile * 16;
  for (int i = lane * 4; i < 16 * 2 * H; i += 128) {
    const float4 v = *(const float4*)(rep + g0 * 2 * H + i);
    *(unsigned long long*)(&Arep[w][i]) = pack4bf(v.x, v.y, v.z, v.w);
  }
  __syncthreads();
  const int crow = (lane >> 4) << 3, col = lane & 15;
#pragma unroll
  for (int nt = 0; nt < 6; ++nt) {                       // hidden 96 cols
    v8f acc; const float bias = b1[nt * 16 + col];
#pragma unroll
    for (int r = 0; r < 8; ++r) acc[r] = bias;
#pragma unroll
    for (int kc = 0; kc < 6; ++kc) {                     // K = 192
      v16bf a = load_a_frag(&Arep[w][0], 2 * H, kc * 32, lane);
      v16bf b = load_b_frag_sw(W1s, 6, kc, nt, lane);
      acc = wmma_bf16(a, b, acc);
    }
#pragma unroll
    for (int r = 0; r < 8; ++r)
      Amid[w][(r + crow) * H + nt * 16 + col] = f2bf(fmaxf(acc[r], 0.0f));
  }
  __syncthreads();
#pragma unroll
  for (int nt = 0; nt < 16; ++nt) {                      // OUT = 256 cols
    v8f acc; const float bias = b2[nt * 16 + col];
#pragma unroll
    for (int r = 0; r < 8; ++r) acc[r] = bias;
#pragma unroll
    for (int kc = 0; kc < 3; ++kc) {                     // K = 96
      v16bf a = load_a_frag(&Amid[w][0], H, kc * 32, lane);
      v16bf b = load_b_frag_g(W2, OUT_D, kc * 32, nt * 16, lane);
      acc = wmma_bf16(a, b, acc);
    }
#pragma unroll
    for (int r = 0; r < 8; ++r)
      out[(g0 + r + crow) * OUT_D + nt * 16 + col] = acc[r];
  }
}

// =============================== host launcher ===============================
extern "C" void kernel_launch(void* const* d_in, const int* in_sizes, int n_in,
                              void* d_out, int out_size, void* d_ws, size_t ws_size,
                              hipStream_t stream) {
  (void)in_sizes; (void)n_in; (void)out_size; (void)ws_size;
  const float* x     = (const float*)d_in[0];
  const int*   ei    = (const int*)d_in[1];
  const float* eattr = (const float*)d_in[2];
  const int*   batch = (const int*)d_in[3];
  const float* nodeW = (const float*)d_in[4];
  const float* nodeb = (const float*)d_in[5];
  const float* edgeW = (const float*)d_in[6];
  const float* edgeb = (const float*)d_in[7];
  const float* epsv  = (const float*)d_in[8];
  const float* cW1   = (const float*)d_in[9];
  const float* cb1   = (const float*)d_in[10];
  const float* cW2   = (const float*)d_in[11];
  const float* cb2   = (const float*)d_in[12];
  const float* gnw   = (const float*)d_in[13];
  const float* gnb   = (const float*)d_in[14];
  const float* gna   = (const float*)d_in[15];
  const float* lnw   = (const float*)d_in[16];
  const float* lnb   = (const float*)d_in[17];
  const float* pW1   = (const float*)d_in[18];
  const float* pb1   = (const float*)d_in[19];
  const float* pW2   = (const float*)d_in[20];
  const float* pb2   = (const float*)d_in[21];
  float* out = (float*)d_out;

  char* base = (char*)d_ws;
  size_t off = 0;
  auto wsalloc = [&](size_t bytes) { void* p = base + off; off = (off + bytes + 255) & ~(size_t)255; return p; };
  float*        h     = (float*)wsalloc((size_t)N_NODES * H * 4);
  float*        t     = (float*)wsalloc((size_t)N_NODES * H * 4);
  float*        agg   = (float*)wsalloc((size_t)N_NODES * H * 4);
  float*        cnt   = (float*)wsalloc((size_t)N_GRAPH * 4);
  float*        inv   = (float*)wsalloc((size_t)N_GRAPH * 4);
  float*        mean  = (float*)wsalloc((size_t)N_GRAPH * H * 4);
  float*        var   = (float*)wsalloc((size_t)N_GRAPH * H * 4);
  float*        gmean = (float*)wsalloc((size_t)N_GRAPH * H * 4);
  unsigned int* gmaxb = (unsigned int*)wsalloc((size_t)N_GRAPH * H * 4);
  float*        rep   = (float*)wsalloc((size_t)N_GRAPH * 2 * H * 4);

  hipMemsetAsync(cnt, 0, (size_t)N_GRAPH * 4, stream);
  k_count<<<N_NODES / 256, 256, 0, stream>>>(batch, cnt);
  k_inv<<<N_GRAPH / 256, 256, 0, stream>>>(cnt, inv);

  k_embed<<<512, 256, 0, stream>>>(x, nodeW, nodeb, h, N_NODES / 16);

  const int nh4_blocks = (N_NODES * (H / 4)) / 256;   // 24576
  for (int l = 0; l < NLAYER; ++l) {
    hipMemsetAsync(agg, 0, (size_t)N_NODES * H * 4, stream);
    k_edge<<<(N_EDGES * (H / 4)) / 256, 256, 0, stream>>>(h, eattr, ei, ei + N_EDGES, edgeW, edgeb, agg);
    k_mlp<<<1024, 128, 0, stream>>>(h, agg, cW1 + l * H * H, cb1 + l * H,
                                    cW2 + l * H * H, cb2 + l * H, epsv, l, t, N_NODES / 16);
    hipMemsetAsync(mean, 0, (size_t)N_GRAPH * H * 4, stream);
    hipMemsetAsync(var, 0, (size_t)N_GRAPH * H * 4, stream);
    k_gsum<<<nh4_blocks, 256, 0, stream>>>(t, batch, mean);
    k_gscale<<<(N_GRAPH * H) / 256, 256, 0, stream>>>(mean, inv);
    k_gvar<<<nh4_blocks, 256, 0, stream>>>(t, batch, mean, gna + l * H, var);
    k_gapply<<<nh4_blocks, 256, 0, stream>>>(t, batch, mean, var, inv,
                                             gnw + l * H, gnb + l * H, gna + l * H, h);
  }

  hipMemsetAsync(gmean, 0, (size_t)N_GRAPH * H * 4, stream);
  hipMemsetAsync(gmaxb, 0, (size_t)N_GRAPH * H * 4, stream);
  k_pool<<<nh4_blocks, 256, 0, stream>>>(h, batch, gmean, gmaxb);
  k_rep_ln<<<N_GRAPH, 192, 0, stream>>>(gmean, gmaxb, inv, lnw, lnb, rep);
  k_head<<<N_GRAPH / 16 / HEAD_WAVES, 32 * HEAD_WAVES, 0, stream>>>(rep, pW1, pb1, pW2, pb2, out);
}
